// Decoder_8194797601151
// MI455X (gfx1250) — matrix operand order repaired and checked by reading, not compiled, not run on hardware
//
#include <hip/hip_runtime.h>
#include <hip/hip_bf16.h>

#define L_SEQ 2048
#define E_DIM 1024
#define H_DIM 1024
#define D_DIM 512
#define V_DIM 32000
#define MV_DIM 32768

typedef __attribute__((ext_vector_type(16))) __bf16 v16bf;
typedef __attribute__((ext_vector_type(8)))  float  v8f;

__device__ __forceinline__ unsigned short f2bf(float f) {
    unsigned u = __builtin_bit_cast(unsigned, f);
    u = u + 0x7FFFu + ((u >> 16) & 1u);   // round-to-nearest-even
    return (unsigned short)(u >> 16);
}

__device__ __forceinline__ float wave_sum(float s) {
#pragma unroll
    for (int m = 16; m >= 1; m >>= 1) s += __shfl_xor(s, m, 32);
    return s;
}

// ---------------------------------------------------------------------------
// Convert EO (2048x1024 f32) -> bf16 A-fragment tiles, order (ib,kb), lane-contiguous.
// A 16x32 bf16 layout: lanes 0-15 row=lane, K {0..7,16..23}; lanes 16-31 row=lane-16, K {8..15,24..31}
// ---------------------------------------------------------------------------
__global__ void convA_k(const float* __restrict__ eo, unsigned short* __restrict__ A) {
    int tid  = blockIdx.x * blockDim.x + threadIdx.x;   // 131072
    int lane = tid & 31;
    int tile = tid >> 5;                                // 0..4095 = ib*32+kb
    int ib = tile >> 5, kb = tile & 31;
    int row   = ib * 16 + (lane & 15);
    int kbase = kb * 32 + ((lane & 16) ? 8 : 0);
    const float* src = eo + (size_t)row * E_DIM + kbase;
    unsigned short h[16];
#pragma unroll
    for (int j = 0; j < 8; ++j) h[j]     = f2bf(src[j]);
#pragma unroll
    for (int j = 0; j < 8; ++j) h[8 + j] = f2bf(src[16 + j]);
    unsigned w[8];
#pragma unroll
    for (int q = 0; q < 8; ++q) w[q] = (unsigned)h[2*q] | ((unsigned)h[2*q+1] << 16);
    uint4* dst = (uint4*)(A + ((size_t)tile * 32 + lane) * 16);
    dst[0] = make_uint4(w[0], w[1], w[2], w[3]);
    dst[1] = make_uint4(w[4], w[5], w[6], w[7]);
}

// ---------------------------------------------------------------------------
// Convert Wc_W (1024x1024 f32, B[k,n]=Wc_W[n,k]) -> bf16 B-fragment tiles,
// order (kb,nb) so one K-step's tiles for a col-group are contiguous (LDS staging).
// B 32x16 bf16 layout: lanes 0-15 col=lane K {0..7,16..23}; lanes 16-31 col=lane-16 K {8..15,24..31}
// ---------------------------------------------------------------------------
__global__ void convB_k(const float* __restrict__ wc, unsigned short* __restrict__ B) {
    int tid  = blockIdx.x * blockDim.x + threadIdx.x;   // 65536
    int lane = tid & 31;
    int tile = tid >> 5;                                // 0..2047 = nb*32+kb (input numbering)
    int nb = tile >> 5, kb = tile & 31;
    int n      = nb * 16 + (lane & 15);
    int kstart = kb * 32 + ((lane & 16) ? 8 : 0);
    const float* src = wc + (size_t)n * E_DIM + kstart; // 2 runs of 8 floats
    unsigned short h[16];
#pragma unroll
    for (int j = 0; j < 16; ++j) h[j] = f2bf(src[(j < 8) ? j : (j + 8)]);
    unsigned w[8];
#pragma unroll
    for (int q = 0; q < 8; ++q) w[q] = (unsigned)h[2*q] | ((unsigned)h[2*q+1] << 16);
    int tileOut = kb * 64 + nb;                         // (kb, nb) ordering
    uint4* dst = (uint4*)(B + ((size_t)tileOut * 32 + lane) * 16);
    dst[0] = make_uint4(w[0], w[1], w[2], w[3]);
    dst[1] = make_uint4(w[4], w[5], w[6], w[7]);
}

// ---------------------------------------------------------------------------
// embedded gather + sel_reading (int matvec over selection mask)
// ---------------------------------------------------------------------------
__global__ void prep_k(const int* __restrict__ inp, const float* __restrict__ emb,
                       const int* __restrict__ seq, const int* __restrict__ pp,
                       const float* __restrict__ eo,
                       float* __restrict__ embedded, float* __restrict__ selread) {
    int tid = blockIdx.x * blockDim.x + threadIdx.x;    // 1536
    int tok = inp[0];
    if (tid < D_DIM) {
        int idx = (tok >= V_DIM) ? 2 : tok;
        embedded[tid] = emb[(size_t)idx * D_DIM + tid];
    } else {
        int e = tid - D_DIM;                            // 0..1023
        int acc = 0;
        for (int i = 0; i < L_SEQ; ++i) {
            int sw = (seq[i] == tok) ? pp[i] : 0;
            if (sw) acc += sw * (int)eo[(size_t)i * E_DIM + e];
        }
        selread[e] = (float)acc;
    }
}

// wave-per-row: attn_W (2048 x 1536) @ concat(embedded, h)
__global__ void attn_logits_k(const float* __restrict__ W, const float* __restrict__ b,
                              const float* __restrict__ embedded, const float* __restrict__ hidden,
                              float* __restrict__ logits) {
    int tid = blockIdx.x * 256 + threadIdx.x;
    int row = tid >> 5, lane = tid & 31;
    const float4* Wr = (const float4*)(W + (size_t)row * (H_DIM + D_DIM));
    float s = 0.f;
#pragma unroll
    for (int k = 0; k < 12; ++k) {
        int c = lane + 32 * k;                          // 0..383
        float4 wv = Wr[c];
        float4 xv = (c < 128) ? ((const float4*)embedded)[c]
                              : ((const float4*)hidden)[c - 128];
        s += wv.x*xv.x + wv.y*xv.y + wv.z*xv.z + wv.w*xv.w;
    }
    s = wave_sum(s);
    if (lane == 0) logits[row] = s + b[row];
}

__global__ void attn_softmax_k(const float* __restrict__ logits, float* __restrict__ attn_out) {
    __shared__ float sm[1024];
    int t = threadIdx.x;
    float a = logits[t], c = logits[t + 1024];
    sm[t] = fmaxf(a, c); __syncthreads();
    for (int s = 512; s > 0; s >>= 1) { if (t < s) sm[t] = fmaxf(sm[t], sm[t + s]); __syncthreads(); }
    float gmx = sm[0]; __syncthreads();
    float ea = __expf(a - gmx), ec = __expf(c - gmx);
    sm[t] = ea + ec; __syncthreads();
    for (int s = 512; s > 0; s >>= 1) { if (t < s) sm[t] += sm[t + s]; __syncthreads(); }
    float inv = 1.0f / sm[0];
    attn_out[t] = ea * inv;
    attn_out[t + 1024] = ec * inv;
}

__global__ void attn_apply_k(const float* __restrict__ w, const float* __restrict__ eo,
                             float* __restrict__ attnapp) {
    int e = blockIdx.x * blockDim.x + threadIdx.x;      // 1024
    float acc = 0.f;
    for (int i = 0; i < L_SEQ; ++i) acc += w[i] * eo[(size_t)i * E_DIM + e];
    attnapp[e] = acc;
}

// wave-per-row: comb_W (1024 x 1536) @ concat(embedded, attn_applied), relu
__global__ void comb_k(const float* __restrict__ W, const float* __restrict__ b,
                       const float* __restrict__ embedded, const float* __restrict__ attnapp,
                       float* __restrict__ outv) {
    int tid = blockIdx.x * 256 + threadIdx.x;
    int row = tid >> 5, lane = tid & 31;
    const float4* Wr = (const float4*)(W + (size_t)row * (E_DIM + D_DIM));
    float s = 0.f;
#pragma unroll
    for (int k = 0; k < 12; ++k) {
        int c = lane + 32 * k;
        float4 wv = Wr[c];
        float4 xv = (c < 128) ? ((const float4*)embedded)[c]
                              : ((const float4*)attnapp)[c - 128];
        s += wv.x*xv.x + wv.y*xv.y + wv.z*xv.z + wv.w*xv.w;
    }
    s = wave_sum(s);
    if (lane == 0) outv[row] = fmaxf(s + b[row], 0.f);
}

// gi = W_ih @ x + b_ih ; gh = W_hh @ h + b_hh (6144 wave-rows)
__global__ void gates_k(const float* __restrict__ Wih, const float* __restrict__ Whh,
                        const float* __restrict__ bih, const float* __restrict__ bhh,
                        const float* __restrict__ xvec, const float* __restrict__ hidden,
                        float* __restrict__ gi, float* __restrict__ gh) {
    int tid = blockIdx.x * 256 + threadIdx.x;
    int w = tid >> 5, lane = tid & 31;
    if (w < 3 * H_DIM) {
        const float4* Wr = (const float4*)(Wih + (size_t)w * 2048);
        const float4* X  = (const float4*)xvec;
        float s = 0.f;
#pragma unroll
        for (int k = 0; k < 16; ++k) {
            int c = lane + 32 * k;
            float4 a = Wr[c], x = X[c];
            s += a.x*x.x + a.y*x.y + a.z*x.z + a.w*x.w;
        }
        s = wave_sum(s);
        if (lane == 0) gi[w] = s + bih[w];
    } else {
        int r = w - 3 * H_DIM;
        const float4* Wr = (const float4*)(Whh + (size_t)r * 1024);
        const float4* X  = (const float4*)hidden;
        float s = 0.f;
#pragma unroll
        for (int k = 0; k < 8; ++k) {
            int c = lane + 32 * k;
            float4 a = Wr[c], x = X[c];
            s += a.x*x.x + a.y*x.y + a.z*x.z + a.w*x.w;
        }
        s = wave_sum(s);
        if (lane == 0) gh[r] = s + bhh[r];
    }
}

__global__ void gru_k(const float* __restrict__ gi, const float* __restrict__ gh,
                      const float* __restrict__ hidden, float* __restrict__ hnew) {
    int i = blockIdx.x * blockDim.x + threadIdx.x;      // 1024
    float rg = 1.f / (1.f + __expf(-(gi[i] + gh[i])));
    float zg = 1.f / (1.f + __expf(-(gi[H_DIM + i] + gh[H_DIM + i])));
    float ng = tanhf(gi[2 * H_DIM + i] + rg * gh[2 * H_DIM + i]);
    hnew[i] = (1.f - zg) * ng + zg * hidden[i];
}

// ---------------------------------------------------------------------------
// Fused WMMA GEMM: score_c[i] = sum_h tanh( (EO @ Wc^T)[i,h] + b[h] ) * h_new[h]
// Block = 8 waves sharing one col-group (64 cols), each wave owns a 16-row block.
// B tiles (4KB per K-step) are staged in LDS with GLOBAL_LOAD_ASYNC_TO_LDS_B128
// (ASYNCcnt path: no VGPR round-trip, overlaps the WMMAs); A is register
// double-buffered with two named fragments so no cross-iteration v_mov copies.
// ---------------------------------------------------------------------------
__global__ void gemm_scorec_k(const unsigned short* __restrict__ A,
                              const unsigned short* __restrict__ B,
                              const float* __restrict__ wcb,
                              const float* __restrict__ hnew,
                              float* __restrict__ part) {
    __shared__ __align__(32) unsigned short smemB[2][2048];   // 2 x 4KB

    int t    = threadIdx.x;                 // 0..255
    int lane = t & 31;
    int wv   = t >> 5;                      // wave in block 0..7
    int cg   = blockIdx.x & 15;             // col group (64 cols)
    int ib   = (blockIdx.x >> 4) * 8 + wv;  // row block 0..127

    // A fragment stream for this wave (kb stride = 1KB = 32 v16bf)
    const v16bf* pA = (const v16bf*)(A + ((size_t)ib * 32 * 32 + lane) * 16);
    // B: per K-step this col-group's chunk is 4KB contiguous at (kb*64 + cg*4)*1KB.
    // Each wave async-copies its 512B slice (32 lanes x 16B).
    unsigned long long gBlane = (unsigned long long)(const char*)B
                              + (unsigned long long)(cg * 4) * 1024
                              + (unsigned)(wv * 512 + lane * 16);
    // low 32 bits of a flat __shared__ pointer == LDS byte offset
    unsigned ldsLane = (unsigned)(size_t)&smemB[0][0] + (unsigned)(wv * 512 + lane * 16);

#define STAGE_B(KB, NBUF)                                                      \
    do {                                                                       \
        unsigned _lds = ldsLane + (unsigned)(NBUF) * 4096u;                    \
        unsigned long long _ga = gBlane + (unsigned long long)(KB) * 65536ull; \
        asm volatile("global_load_async_to_lds_b128 %0, %1, off"               \
                     :: "v"(_lds), "v"(_ga) : "memory");                       \
    } while (0)

    v8f acc0 = {0,0,0,0,0,0,0,0};
    v8f acc1 = {0,0,0,0,0,0,0,0};
    v8f acc2 = {0,0,0,0,0,0,0,0};
    v8f acc3 = {0,0,0,0,0,0,0,0};

    // prologue: async-stage kb=0, preload A frag 0
    STAGE_B(0, 0);
    v16bf a0 = pA[0];
    v16bf a1;
    asm volatile("s_wait_asynccnt 0x0" ::: "memory");
    __syncthreads();

#define GEMM_STEP(KB, CUR, ACUR, ANXT)                                          \
    do {                                                                        \
        if ((KB) < 31) {                                                        \
            STAGE_B((KB) + 1, (CUR) ^ 1);                                       \
            ANXT = pA[((KB) + 1) * 32];                                         \
        }                                                                       \
        const unsigned short* sb = &smemB[CUR][0];                              \
        v16bf b0 = *(const v16bf*)(sb + 0 * 512 + lane * 16);                   \
        v16bf b1 = *(const v16bf*)(sb + 1 * 512 + lane * 16);                   \
        v16bf b2 = *(const v16bf*)(sb + 2 * 512 + lane * 16);                   \
        v16bf b3 = *(const v16bf*)(sb + 3 * 512 + lane * 16);                   \
        acc0 = __builtin_amdgcn_wmma_f32_16x16x32_bf16(false, ACUR, false, b0, (short)0, acc0, false, false); \
        acc1 = __builtin_amdgcn_wmma_f32_16x16x32_bf16(false, ACUR, false, b1, (short)0, acc1, false, false); \
        acc2 = __builtin_amdgcn_wmma_f32_16x16x32_bf16(false, ACUR, false, b2, (short)0, acc2, false, false); \
        acc3 = __builtin_amdgcn_wmma_f32_16x16x32_bf16(false, ACUR, false, b3, (short)0, acc3, false, false); \
        if ((KB) < 31) asm volatile("s_wait_asynccnt 0x0" ::: "memory");        \
        __syncthreads();                                                        \
    } while (0)

#pragma unroll 1
    for (int k2 = 0; k2 < 16; ++k2) {
        int kb = 2 * k2;
        GEMM_STEP(kb,     0, a0, a1);
        GEMM_STEP(kb + 1, 1, a1, a0);
    }
#undef GEMM_STEP
#undef STAGE_B

    // epilogue: tanh(+bias) * h_new[col], reduce 16 cols per row half
    int nlo = lane & 15;
    int n0  = cg * 64 + nlo;
    float bv0 = wcb[n0],      hv0 = hnew[n0];
    float bv1 = wcb[n0 + 16], hv1 = hnew[n0 + 16];
    float bv2 = wcb[n0 + 32], hv2 = hnew[n0 + 32];
    float bv3 = wcb[n0 + 48], hv3 = hnew[n0 + 48];
    int rowbase = ib * 16 + ((lane & 16) ? 8 : 0);
#pragma unroll
    for (int r = 0; r < 8; ++r) {
        float s = tanhf(acc0[r] + bv0) * hv0
                + tanhf(acc1[r] + bv1) * hv1
                + tanhf(acc2[r] + bv2) * hv2
                + tanhf(acc3[r] + bv3) * hv3;
#pragma unroll
        for (int m = 8; m >= 1; m >>= 1) s += __shfl_xor(s, m, 32);  // within 16-lane half
        if (nlo == 0) part[(size_t)cg * L_SEQ + rowbase + r] = s;    // deterministic partials
    }
}

__global__ void scorec_fin_k(const float* __restrict__ part, float* __restrict__ score) {
    int i = blockIdx.x * blockDim.x + threadIdx.x;      // 2048
    float s = 0.f;
#pragma unroll
    for (int cg = 0; cg < 16; ++cg) s += part[(size_t)cg * L_SEQ + i];
    score[V_DIM + i] = s;
}

// wave-per-row: Wo_W (32000 x 1024) @ h_new  (131 MB -> dominates HBM time)
__global__ void scoreg_k(const float* __restrict__ W, const float* __restrict__ b,
                         const float* __restrict__ hnew, float* __restrict__ score) {
    int tid = blockIdx.x * 256 + threadIdx.x;
    int row = tid >> 5, lane = tid & 31;
    const float4* Wr = (const float4*)(W + (size_t)row * 1024);
    const float4* X  = (const float4*)hnew;
    float s = 0.f;
#pragma unroll
    for (int k = 0; k < 8; ++k) {
        int c = lane + 32 * k;
        float4 a = Wr[c], x = X[c];
        s += a.x*x.x + a.y*x.y + a.z*x.z + a.w*x.w;
    }
    s = wave_sum(s);
    if (lane == 0) score[row] = s + b[row];
}

__global__ void score_reduce_k(const float* __restrict__ score, float* __restrict__ red) {
    __shared__ float sm[1024];
    int t = threadIdx.x;
    float mx = -3.4e38f;
    for (int j = t; j < V_DIM + L_SEQ; j += 1024) mx = fmaxf(mx, score[j]);
    sm[t] = mx; __syncthreads();
    for (int s = 512; s > 0; s >>= 1) { if (t < s) sm[t] = fmaxf(sm[t], sm[t + s]); __syncthreads(); }
    float gmx = sm[0]; __syncthreads();
    float sum = 0.f;
    for (int j = t; j < V_DIM + L_SEQ; j += 1024) sum += __expf(score[j] - gmx);
    sm[t] = sum; __syncthreads();
    for (int s = 512; s > 0; s >>= 1) { if (t < s) sm[t] += sm[t + s]; __syncthreads(); }
    if (t == 0) { red[0] = gmx; red[1] = sm[0]; }
}

__global__ void outA_k(const float* __restrict__ score, const float* __restrict__ red,
                       float* __restrict__ out1) {
    int j = blockIdx.x * blockDim.x + threadIdx.x;      // MV
    float p = 0.f;
    if (j < V_DIM) p = __expf(score[j] - red[0]) / red[1];
    out1[j] = p;
}

__global__ void outB_k(const float* __restrict__ score, const float* __restrict__ red,
                       const int* __restrict__ seq, float* __restrict__ out1,
                       float* __restrict__ probc) {
    int i = blockIdx.x * blockDim.x + threadIdx.x;      // L
    float pc = __expf(score[V_DIM + i] - red[0]) / red[1];
    probc[i] = pc;
    atomicAdd(&out1[seq[i]], pc);
}

__global__ void outC_k(float* __restrict__ out1) {
    int j = blockIdx.x * blockDim.x + threadIdx.x;      // MV
    float v = out1[j];
    if (j == 2) v = 1e-9f;
    else if (v == 0.f) v = 1e-9f;
    out1[j] = __logf(v);
}

extern "C" void kernel_launch(void* const* d_in, const int* in_sizes, int n_in,
                              void* d_out, int out_size, void* d_ws, size_t ws_size,
                              hipStream_t stream) {
    const int*   inp    = (const int*)d_in[0];
    const float* hidden = (const float*)d_in[1];
    const float* eo     = (const float*)d_in[2];
    const int*   seq    = (const int*)d_in[3];
    const int*   pp     = (const int*)d_in[4];
    const float* emb    = (const float*)d_in[5];
    const float* attnW  = (const float*)d_in[6];
    const float* attnb  = (const float*)d_in[7];
    const float* combW  = (const float*)d_in[8];
    const float* combb  = (const float*)d_in[9];
    const float* Wih    = (const float*)d_in[10];
    const float* Whh    = (const float*)d_in[11];
    const float* bih    = (const float*)d_in[12];
    const float* bhh    = (const float*)d_in[13];
    const float* WoW    = (const float*)d_in[14];
    const float* Wob    = (const float*)d_in[15];
    const float* WcW    = (const float*)d_in[16];
    const float* Wcb    = (const float*)d_in[17];

    float* out1     = (float*)d_out;            // [0, 32768)
    float* hnew_out = out1 + MV_DIM;            // [32768, 33792)
    float* attn_out = hnew_out + H_DIM;         // [33792, 35840)
    float* probc    = attn_out + L_SEQ;         // [35840, 37888)

    // ws: 4MB bf16 A tiles | 2MB bf16 B tiles | scalars (~310KB)
    unsigned short* wsA = (unsigned short*)d_ws;
    unsigned short* wsB = (unsigned short*)((char*)d_ws + (4u << 20));
    float* wf       = (float*)((char*)d_ws + (6u << 20));
    float* embedded = wf;                       // 512
    float* xvec     = wf + 512;                 // 2048 = [out | sel_reading]
    float* selread  = xvec + 1024;
    float* logits   = wf + 2560;                // 2048 (reused as attn_applied)
    float* attnapp  = logits;
    float* gi       = wf + 4608;                // 3072
    float* gh       = wf + 7680;                // 3072
    float* score    = wf + 10752;               // 34048 = [score_g | score_c]
    float* part     = wf + 44800;               // 16 x 2048 GEMM partials
    float* red      = wf + 77568;               // max, sumexp

    convA_k<<<512, 256, 0, stream>>>(eo, wsA);
    convB_k<<<256, 256, 0, stream>>>(WcW, wsB);
    prep_k<<<6, 256, 0, stream>>>(inp, emb, seq, pp, eo, embedded, selread);
    attn_logits_k<<<256, 256, 0, stream>>>(attnW, attnb, embedded, hidden, logits);
    attn_softmax_k<<<1, 1024, 0, stream>>>(logits, attn_out);
    attn_apply_k<<<4, 256, 0, stream>>>(attn_out, eo, attnapp);
    comb_k<<<128, 256, 0, stream>>>(combW, combb, embedded, attnapp, xvec);
    gates_k<<<768, 256, 0, stream>>>(Wih, Whh, bih, bhh, xvec, hidden, gi, gh);
    gru_k<<<4, 256, 0, stream>>>(gi, gh, hidden, hnew_out);
    scoreg_k<<<4000, 256, 0, stream>>>(WoW, Wob, hnew_out, score);
    gemm_scorec_k<<<256, 256, 0, stream>>>(wsA, wsB, Wcb, hnew_out, part);
    scorec_fin_k<<<8, 256, 0, stream>>>(part, score);
    score_reduce_k<<<1, 1024, 0, stream>>>(score, red);
    outA_k<<<128, 256, 0, stream>>>(score, red, out1);
    outB_k<<<8, 256, 0, stream>>>(score, red, seq, out1, probc);
    outC_k<<<128, 256, 0, stream>>>(out1);
}
